// GPT_1743756722334
// MI455X (gfx1250) — compile-verified, matrix operand order checked by reading, not run on hardware
//
#include <hip/hip_runtime.h>
#include <math.h>

// GPT-2 small forward: D=1024, H=16, L=12, V=50257, B=4, T=1024
#define B_    4
#define T_    1024
#define D_    1024
#define H_    16
#define HD_   64
#define L_    12
#define V_    50257
#define MTOT  (B_ * T_)          // 4096 rows

typedef __attribute__((ext_vector_type(16))) __bf16 v16bf;
typedef __attribute__((ext_vector_type(8)))  float  v8f;

union Frag16 {                    // 16 bf16 = 8 VGPRs (wave32 WMMA A/B operand)
    v16bf v;
    uint4 q[2];
};

static __device__ __forceinline__ v8f wmma_bf16(const Frag16& a, const Frag16& b, v8f c)
{
    return __builtin_amdgcn_wmma_f32_16x16x32_bf16(false, a.v, false, b.v,
                                                   (short)0, c, false, false);
}

// ---------------------------------------------------------------------------
// Embedding: x[b,t,:] = wte[idx[b,t],:] + wpe[t,:]
// ---------------------------------------------------------------------------
__global__ __launch_bounds__(256) void embed_kernel(
    const int* __restrict__ idx, const float* __restrict__ wte,
    const float* __restrict__ wpe, float* __restrict__ x)
{
    int bt  = blockIdx.x;
    int t   = bt & (T_ - 1);
    int tok = idx[bt];
    int tid = threadIdx.x;                       // 256 threads * float4 = 1024
    float4 a = ((const float4*)(wte + (size_t)tok * D_))[tid];
    float4 p = ((const float4*)(wpe + (size_t)t   * D_))[tid];
    float4 o; o.x = a.x + p.x; o.y = a.y + p.y; o.z = a.z + p.z; o.w = a.w + p.w;
    ((float4*)(x + (size_t)bt * D_))[tid] = o;
}

// ---------------------------------------------------------------------------
// LayerNorm: one block per row, D=1024, fp32 reductions
// ---------------------------------------------------------------------------
__global__ __launch_bounds__(256) void ln_kernel(
    const float* __restrict__ x, const float* __restrict__ w,
    const float* __restrict__ b, float* __restrict__ y)
{
    int row = blockIdx.x;
    int tid = threadIdx.x;
    float4 v = ((const float4*)(x + (size_t)row * D_))[tid];
    float s  = v.x + v.y + v.z + v.w;
    float ss = v.x*v.x + v.y*v.y + v.z*v.z + v.w*v.w;
    #pragma unroll
    for (int o = 16; o > 0; o >>= 1) {
        s  += __shfl_xor(s,  o, 32);
        ss += __shfl_xor(ss, o, 32);
    }
    __shared__ float rs[8], rss[8];
    int wave = tid >> 5, lane = tid & 31;
    if (lane == 0) { rs[wave] = s; rss[wave] = ss; }
    __syncthreads();
    float st = 0.f, sst = 0.f;
    #pragma unroll
    for (int i = 0; i < 8; ++i) { st += rs[i]; sst += rss[i]; }
    float mean = st * (1.0f / D_);
    float var  = sst * (1.0f / D_) - mean * mean;
    float rstd = rsqrtf(var + 1e-5f);
    float4 wv = ((const float4*)w)[tid];
    float4 bv = ((const float4*)b)[tid];
    float4 o;
    o.x = (v.x - mean) * rstd * wv.x + bv.x;
    o.y = (v.y - mean) * rstd * wv.y + bv.y;
    o.z = (v.z - mean) * rstd * wv.z + bv.z;
    o.w = (v.w - mean) * rstd * wv.w + bv.w;
    ((float4*)(y + (size_t)row * D_))[tid] = o;
}

// ---------------------------------------------------------------------------
// Tiled GEMM, bf16 WMMA, fp32 accumulate, register-prefetch + LDS double buffer
// + global_prefetch of the weight stream two K-tiles ahead.
//   C[M,N] = A[M,K] @ B + bias (+ residual) (+ GELU)
// TRANSB=0: B is [K,N] row-major (stride N), N % 128 == 0 guaranteed.
// TRANSB=1: B is [N,K] row-major (stride ldb), ragged N allowed (LM head).
// Tile 128x128x32; 8 waves; each wave: 2(M) x 4(N) v_wmma_f32_16x16x32_bf16.
// ---------------------------------------------------------------------------
template <bool TRANSB, bool GELU, bool RES>
__global__ __launch_bounds__(256) void gemm_kernel(
    const float* __restrict__ A, const float* __restrict__ Bm,
    const float* __restrict__ bias, const float* __restrict__ Rsd,
    float* __restrict__ C, int M, int N, int K, int ldb)
{
    __shared__ __align__(16) __bf16 lA[2][128 * 32];   // [m][k], 8 KB each
    __shared__ __align__(16) __bf16 lB[2][128 * 32];   // [n][k], 8 KB each

    const int tid  = threadIdx.x;
    const int bm   = blockIdx.y * 128;
    const int bn   = blockIdx.x * 128;
    const int lane = tid & 31;
    const int wave = tid >> 5;
    const int wm   = (wave & 3) * 32;    // 4 waves along M
    const int wn   = (wave >> 2) * 64;   // 2 waves along N
    const int kh   = (lane >> 4) * 8;    // K-half select per ISA operand layout
    const int lm   = lane & 15;

    float4 ra[4], rb[4];

    auto loadA = [&](int k0) {
        #pragma unroll
        for (int it = 0; it < 4; ++it) {
            int id = tid + it * 256;
            int r  = id >> 3;
            int c  = (id & 7) << 2;
            ra[it] = *(const float4*)(A + (size_t)(bm + r) * K + k0 + c);
        }
    };
    auto loadB = [&](int k0) {
        if (TRANSB) {
            #pragma unroll
            for (int it = 0; it < 4; ++it) {
                int id = tid + it * 256;
                int r  = id >> 3;           // n in tile
                int c  = (id & 7) << 2;     // k in tile
                int n  = bn + r;
                if (n < N) rb[it] = *(const float4*)(Bm + (size_t)n * ldb + k0 + c);
                else       { rb[it].x = rb[it].y = rb[it].z = rb[it].w = 0.f; }
            }
        } else {
            #pragma unroll
            for (int it = 0; it < 4; ++it) {
                int id = tid + it * 256;
                int r  = id >> 5;           // k in tile 0..31
                int c  = (id & 31) << 2;    // n in tile (N % 128 == 0 here)
                rb[it] = *(const float4*)(Bm + (size_t)(k0 + r) * N + bn + c);
            }
        }
    };
    auto storeTiles = [&](int buf) {
        #pragma unroll
        for (int it = 0; it < 4; ++it) {
            int id = tid + it * 256;
            int r  = id >> 3;
            int c  = (id & 7) << 2;
            alignas(8) __bf16 h[4] = {(__bf16)ra[it].x, (__bf16)ra[it].y,
                                      (__bf16)ra[it].z, (__bf16)ra[it].w};
            *(uint2*)(&lA[buf][r * 32 + c]) = *(const uint2*)h;
        }
        if (TRANSB) {
            #pragma unroll
            for (int it = 0; it < 4; ++it) {
                int id = tid + it * 256;
                int r  = id >> 3;
                int c  = (id & 7) << 2;
                alignas(8) __bf16 h[4] = {(__bf16)rb[it].x, (__bf16)rb[it].y,
                                          (__bf16)rb[it].z, (__bf16)rb[it].w};
                *(uint2*)(&lB[buf][r * 32 + c]) = *(const uint2*)h;
            }
        } else {
            #pragma unroll
            for (int it = 0; it < 4; ++it) {
                int id = tid + it * 256;
                int r  = id >> 5;
                int c  = (id & 31) << 2;
                lB[buf][(c + 0) * 32 + r] = (__bf16)rb[it].x;
                lB[buf][(c + 1) * 32 + r] = (__bf16)rb[it].y;
                lB[buf][(c + 2) * 32 + r] = (__bf16)rb[it].z;
                lB[buf][(c + 3) * 32 + r] = (__bf16)rb[it].w;
            }
        }
    };

    v8f acc[2][4];
    const v8f vzero = {0.f,0.f,0.f,0.f,0.f,0.f,0.f,0.f};
    #pragma unroll
    for (int i = 0; i < 2; ++i)
        #pragma unroll
        for (int j = 0; j < 4; ++j) acc[i][j] = vzero;

    const int nt = K >> 5;
    loadA(0); loadB(0);
    storeTiles(0);

    for (int i = 0; i < nt; ++i) {
        __syncthreads();
        if (i + 1 < nt) { loadA((i + 1) << 5); loadB((i + 1) << 5); }  // prefetch next tile
        if (i + 2 < nt) {
            // stream weights into cache two K-tiles ahead (global_prefetch_b8)
            const int k0 = (i + 2) << 5;
            if (TRANSB) {
                int n = bn + (tid >> 3);
                if (n < N)
                    __builtin_prefetch(Bm + (size_t)n * ldb + k0 + ((tid & 7) << 2), 0, 0);
            } else {
                __builtin_prefetch(Bm + (size_t)(k0 + (tid >> 5)) * N + bn + ((tid & 31) << 2), 0, 0);
            }
        }
        const int buf = i & 1;

        Frag16 af[2], bf[4];
        #pragma unroll
        for (int im = 0; im < 2; ++im) {
            const __bf16* p = &lA[buf][(wm + im * 16 + lm) * 32 + kh];
            af[im].q[0] = *(const uint4*)(p);        // K = kh .. kh+7
            af[im].q[1] = *(const uint4*)(p + 16);   // K = kh+16 .. kh+23
        }
        #pragma unroll
        for (int in = 0; in < 4; ++in) {
            const __bf16* p = &lB[buf][(wn + in * 16 + lm) * 32 + kh];
            bf[in].q[0] = *(const uint4*)(p);
            bf[in].q[1] = *(const uint4*)(p + 16);
        }
        #pragma unroll
        for (int im = 0; im < 2; ++im)
            #pragma unroll
            for (int in = 0; in < 4; ++in)
                acc[im][in] = wmma_bf16(af[im], bf[in], acc[im][in]);

        if (i + 1 < nt) storeTiles(buf ^ 1);   // waits on prefetch, fills other buffer
    }

    // ---- epilogue: bias (+residual) (+gelu) ----
    #pragma unroll
    for (int im = 0; im < 2; ++im) {
        int rbase = bm + wm + im * 16 + (lane >> 4) * 8;   // C layout M rows
        #pragma unroll
        for (int in = 0; in < 4; ++in) {
            int col = bn + wn + in * 16 + lm;
            if (col < N) {
                float bv = bias[col];
                #pragma unroll
                for (int i = 0; i < 8; ++i) {
                    int row = rbase + i;
                    float v = acc[im][in][i] + bv;
                    if (RES)  v += Rsd[(size_t)row * N + col];
                    if (GELU) v = 0.5f * v * (1.0f + erff(v * 0.70710678118654752f));
                    C[(size_t)row * N + col] = v;
                }
            }
        }
    }
}

// ---------------------------------------------------------------------------
// fp32 -> bf16 bulk convert (qkv activations)
// ---------------------------------------------------------------------------
__global__ __launch_bounds__(256) void cvt_bf16_kernel(
    const float* __restrict__ in, __bf16* __restrict__ out)
{
    int i = blockIdx.x * 256 + threadIdx.x;      // one float4 -> 4 bf16 each
    float4 f = ((const float4*)in)[i];
    alignas(8) __bf16 h[4] = {(__bf16)f.x, (__bf16)f.y, (__bf16)f.z, (__bf16)f.w};
    ((uint2*)out)[i] = *(const uint2*)h;
}

// ---------------------------------------------------------------------------
// Build V^T in bf16: vt[(b*H+h)*64 + d][j] = qkv[(b*T+j)*3D + 2D + h*64 + d]
// ---------------------------------------------------------------------------
__global__ __launch_bounds__(256) void vt_kernel(
    const float* __restrict__ qkv, __bf16* __restrict__ vt)
{
    int bhd = blockIdx.x;            // (b*16+h)*64 + d
    int d   = bhd & 63;
    int bh  = bhd >> 6;
    int b   = bh >> 4;
    int h   = bh & 15;
    int j4  = threadIdx.x * 4;
    const float* src = qkv + (size_t)(b * T_ + j4) * (3 * D_) + 2 * D_ + h * HD_ + d;
    alignas(8) __bf16 hh[4] = {(__bf16)src[0],
                               (__bf16)src[3 * D_],
                               (__bf16)src[2 * 3 * D_],
                               (__bf16)src[3 * 3 * D_]};
    *(uint2*)(vt + (size_t)bhd * T_ + j4) = *(const uint2*)hh;
}

// ---------------------------------------------------------------------------
// WMMA flash attention. One wave per 16-row Q tile; 32 keys per step.
// qkvh: bf16 [B*T][3D]; vt: bf16 [B*H*64][T]; out: fp32 [B*T][D].
// Causal masking is hoisted out of the hot loop: steps with j0+31 <= q0 need
// no masking (the vast majority across the triangular sweep); only the
// diagonal-straddling tail steps pay the compare/select cost.
// No block barriers (waves diverge); P transpose via per-wave LDS tile with
// explicit s_wait_dscnt (wave-lockstep, LDS ops are in-order per wave).
// ---------------------------------------------------------------------------
__global__ __launch_bounds__(256) void fattn_kernel(
    const __bf16* __restrict__ qkvh, const __bf16* __restrict__ vt,
    float* __restrict__ out)
{
    __shared__ __align__(16) __bf16 pbuf[8][16 * 32];   // per-wave P tile, 1 KB each

    const int wave = threadIdx.x >> 5;
    const int lane = threadIdx.x & 31;
    const int gw   = blockIdx.x * 8 + wave;   // 0 .. B*H*(T/16)-1 = 4095
    const int qt   = gw & 63;                 // q tile (T/16 = 64)
    const int bh   = gw >> 6;                 // b*16 + h
    const int b    = bh >> 4;
    const int h    = bh & 15;
    const int q0   = qt * 16;
    const int lm   = lane & 15;
    const int hi   = lane >> 4;
    const int kh   = hi * 8;
    __bf16* pl = pbuf[wave];

    // Q A-fragments (row m = lm, K = head dim, two 32-wide steps)
    Frag16 aq[2];
    {
        const __bf16* qp = qkvh + (size_t)(b * T_ + q0 + lm) * (3 * D_) + h * HD_;
        #pragma unroll
        for (int s = 0; s < 2; ++s) {
            aq[s].q[0] = *(const uint4*)(qp + s * 32 + kh);
            aq[s].q[1] = *(const uint4*)(qp + s * 32 + kh + 16);
        }
    }

    v8f o[4];
    const v8f vzero = {0.f,0.f,0.f,0.f,0.f,0.f,0.f,0.f};
    #pragma unroll
    for (int t = 0; t < 4; ++t) o[t] = vzero;
    float rm[8], rl[8];
    #pragma unroll
    for (int i = 0; i < 8; ++i) { rm[i] = -3.0e38f; rl[i] = 0.f; }

    auto kvstep = [&](int j0, bool domask) {
        // ---- S = Q @ K^T for 32 keys (two 16-col C tiles) ----
        v8f s0 = vzero, s1 = vzero;
        const __bf16* kp0 = qkvh + (size_t)(b * T_ + j0 + lm) * (3 * D_) + D_ + h * HD_;
        const __bf16* kp1 = kp0 + (size_t)16 * (3 * D_);
        #pragma unroll
        for (int s = 0; s < 2; ++s) {
            Frag16 bk;
            bk.q[0] = *(const uint4*)(kp0 + s * 32 + kh);
            bk.q[1] = *(const uint4*)(kp0 + s * 32 + kh + 16);
            s0 = wmma_bf16(aq[s], bk, s0);
        }
        #pragma unroll
        for (int s = 0; s < 2; ++s) {
            Frag16 bk;
            bk.q[0] = *(const uint4*)(kp1 + s * 32 + kh);
            bk.q[1] = *(const uint4*)(kp1 + s * 32 + kh + 16);
            s1 = wmma_bf16(aq[s], bk, s1);
        }
        // ---- online softmax per row (rows live in c[i], cols stripe 16 lanes) ----
        #pragma unroll
        for (int i = 0; i < 8; ++i) {
            const int row = q0 + i + hi * 8;
            float x0 = s0[i] * 0.125f;
            float x1 = s1[i] * 0.125f;
            if (domask) {
                if (j0 + lm      > row) x0 = -3.0e38f;
                if (j0 + 16 + lm > row) x1 = -3.0e38f;
            }
            float t = fmaxf(x0, x1);
            #pragma unroll
            for (int off = 1; off < 16; off <<= 1) t = fmaxf(t, __shfl_xor(t, off, 32));
            float mn = fmaxf(rm[i], t);
            float c  = __expf(rm[i] - mn);
            float p0 = __expf(x0 - mn);
            float p1 = __expf(x1 - mn);
            float ps = p0 + p1;
            #pragma unroll
            for (int off = 1; off < 16; off <<= 1) ps += __shfl_xor(ps, off, 32);
            rl[i] = rl[i] * c + ps;
            rm[i] = mn;
            o[0][i] *= c; o[1][i] *= c; o[2][i] *= c; o[3][i] *= c;
            pl[(i + hi * 8) * 32 + lm]      = (__bf16)p0;
            pl[(i + hi * 8) * 32 + 16 + lm] = (__bf16)p1;
        }
        // wave-local LDS turnaround: stores complete before transposed reads
        asm volatile("s_wait_dscnt 0x0" ::: "memory");
        Frag16 pa;
        {
            const __bf16* pp = pl + lm * 32 + kh;
            pa.q[0] = *(const uint4*)(pp);
            pa.q[1] = *(const uint4*)(pp + 16);
        }
        // ---- O += P @ V (V^T rows are contiguous in key index) ----
        #pragma unroll
        for (int t = 0; t < 4; ++t) {
            const __bf16* vp = vt + (size_t)(bh * 64 + t * 16 + lm) * T_ + j0;
            Frag16 bv;
            bv.q[0] = *(const uint4*)(vp + kh);
            bv.q[1] = *(const uint4*)(vp + kh + 16);
            o[t] = wmma_bf16(pa, bv, o[t]);
        }
    };

    const int nfull = (q0 + 1) >> 5;          // steps entirely below the diagonal
    const int nst   = (q0 + 16 + 31) >> 5;    // total 32-key steps
    int j0 = 0;
    for (int st = 0; st < nfull; ++st, j0 += 32) kvstep(j0, false);
    for (int st = nfull; st < nst; ++st, j0 += 32) kvstep(j0, true);

    // ---- finalize and write [B*T][D] ----
    #pragma unroll
    for (int i = 0; i < 8; ++i) {
        const float inv = 1.0f / rl[i];
        const int   row = q0 + i + hi * 8;
        float* op = out + (size_t)(b * T_ + row) * D_ + h * HD_;
        #pragma unroll
        for (int t = 0; t < 4; ++t)
            op[t * 16 + lm] = o[t][i] * inv;
    }
}

// ---------------------------------------------------------------------------
// Loss: per-row log-softmax + NLL; mean accumulated atomically.
// ---------------------------------------------------------------------------
__global__ void zero_kernel(float* p) { if (threadIdx.x == 0) *p = 0.f; }

__global__ __launch_bounds__(256) void loss_kernel(
    const float* __restrict__ logits, const int* __restrict__ targets,
    float* __restrict__ loss)
{
    int row = blockIdx.x;
    const float* lr = logits + (size_t)row * V_;
    int tid = threadIdx.x, wave = tid >> 5, lane = tid & 31;

    float m = -3.0e38f;
    for (int i = tid; i < V_; i += 256) m = fmaxf(m, lr[i]);
    #pragma unroll
    for (int o = 16; o > 0; o >>= 1) m = fmaxf(m, __shfl_xor(m, o, 32));
    __shared__ float sm[8], ss[8];
    if (lane == 0) sm[wave] = m;
    __syncthreads();
    float mm = sm[0];
    #pragma unroll
    for (int i = 1; i < 8; ++i) mm = fmaxf(mm, sm[i]);

    float s = 0.f;
    for (int i = tid; i < V_; i += 256) s += expf(lr[i] - mm);
    #pragma unroll
    for (int o = 16; o > 0; o >>= 1) s += __shfl_xor(s, o, 32);
    if (lane == 0) ss[wave] = s;
    __syncthreads();
    if (tid == 0) {
        float st = 0.f;
        #pragma unroll
        for (int i = 0; i < 8; ++i) st += ss[i];
        int t = targets[row];
        float nll = -(lr[t] - mm - logf(st));
        atomicAdd(loss, nll * (1.0f / (float)MTOT));
    }
}

// ---------------------------------------------------------------------------
// Driver
// ---------------------------------------------------------------------------
extern "C" void kernel_launch(void* const* d_in, const int* in_sizes, int n_in,
                              void* d_out, int out_size, void* d_ws, size_t ws_size,
                              hipStream_t stream)
{
    const int*   idx     = (const int*)  d_in[0];
    const int*   targets = (const int*)  d_in[1];
    const float* wte     = (const float*)d_in[2];
    const float* wpe     = (const float*)d_in[3];
    const float* ln1_w   = (const float*)d_in[4];
    const float* ln1_b   = (const float*)d_in[5];
    const float* qkv_w   = (const float*)d_in[6];
    const float* qkv_b   = (const float*)d_in[7];
    const float* o_w     = (const float*)d_in[8];
    const float* o_b     = (const float*)d_in[9];
    const float* ln2_w   = (const float*)d_in[10];
    const float* ln2_b   = (const float*)d_in[11];
    const float* up_w    = (const float*)d_in[12];
    const float* up_b    = (const float*)d_in[13];
    const float* down_w  = (const float*)d_in[14];
    const float* down_b  = (const float*)d_in[15];
    const float* lnf_w   = (const float*)d_in[16];
    const float* lnf_b   = (const float*)d_in[17];
    const float* lm_b    = (const float*)d_in[18];

    // workspace carve-out (192 MB total)
    char*   ws    = (char*)d_ws;
    float*  x     = (float*) (ws);                                  // 16 MB
    float*  lnbuf = (float*) (ws + (size_t)16  * 1024 * 1024);      // 16 MB
    float*  attnb = (float*) (ws + (size_t)32  * 1024 * 1024);      // 16 MB
    float*  qkvb  = (float*) (ws + (size_t)48  * 1024 * 1024);      // 48 MB
    float*  upb   = (float*) (ws + (size_t)96  * 1024 * 1024);      // 64 MB
    __bf16* qkvh  = (__bf16*)(ws + (size_t)160 * 1024 * 1024);      // 24 MB
    __bf16* vtb   = (__bf16*)(ws + (size_t)184 * 1024 * 1024);      //  8 MB

    float* logits = (float*)d_out;
    float* loss   = logits + (size_t)MTOT * V_;

    embed_kernel<<<MTOT, 256, 0, stream>>>(idx, wte, wpe, x);

    for (int l = 0; l < L_; ++l) {
        const float* l1w = ln1_w  + (size_t)l * D_;
        const float* l1b = ln1_b  + (size_t)l * D_;
        const float* qw  = qkv_w  + (size_t)l * D_ * (3 * D_);
        const float* qb  = qkv_b  + (size_t)l * (3 * D_);
        const float* ow  = o_w    + (size_t)l * D_ * D_;
        const float* ob  = o_b    + (size_t)l * D_;
        const float* l2w = ln2_w  + (size_t)l * D_;
        const float* l2b = ln2_b  + (size_t)l * D_;
        const float* uw  = up_w   + (size_t)l * D_ * (4 * D_);
        const float* ub  = up_b   + (size_t)l * (4 * D_);
        const float* dw  = down_w + (size_t)l * (4 * D_) * D_;
        const float* db  = down_b + (size_t)l * D_;

        ln_kernel<<<MTOT, 256, 0, stream>>>(x, l1w, l1b, lnbuf);

        dim3 gq(3 * D_ / 128, MTOT / 128);
        gemm_kernel<false, false, false><<<gq, 256, 0, stream>>>(
            lnbuf, qw, qb, nullptr, qkvb, MTOT, 3 * D_, D_, 0);

        // attention preprocessing + WMMA flash attention
        cvt_bf16_kernel<<<(MTOT * 3 * D_) / (256 * 4), 256, 0, stream>>>(qkvb, qkvh);
        vt_kernel<<<B_ * H_ * HD_, 256, 0, stream>>>(qkvb, vtb);
        fattn_kernel<<<(B_ * H_ * (T_ / 16)) / 8, 256, 0, stream>>>(qkvh, vtb, attnb);

        dim3 go(D_ / 128, MTOT / 128);
        gemm_kernel<false, false, true><<<go, 256, 0, stream>>>(
            attnb, ow, ob, x, x, MTOT, D_, D_, 0);

        ln_kernel<<<MTOT, 256, 0, stream>>>(x, l2w, l2b, lnbuf);

        dim3 gu(4 * D_ / 128, MTOT / 128);
        gemm_kernel<false, true, false><<<gu, 256, 0, stream>>>(
            lnbuf, uw, ub, nullptr, upb, MTOT, 4 * D_, D_, 0);

        dim3 gd(D_ / 128, MTOT / 128);
        gemm_kernel<false, false, true><<<gd, 256, 0, stream>>>(
            upb, dw, db, x, x, MTOT, D_, 4 * D_, 0);
    }

    ln_kernel<<<MTOT, 256, 0, stream>>>(x, lnf_w, lnf_b, lnbuf);

    dim3 gl((V_ + 127) / 128, MTOT / 128);
    gemm_kernel<true, false, false><<<gl, 256, 0, stream>>>(
        lnbuf, wte, lm_b, nullptr, logits, MTOT, V_, D_, D_);

    zero_kernel<<<1, 32, 0, stream>>>(loss);
    loss_kernel<<<MTOT, 256, 0, stream>>>(logits, targets, loss);
}
